// Cell_75969381532224
// MI455X (gfx1250) — compile-verified, matrix operand order checked by reading, not run on hardware
//
#include <hip/hip_runtime.h>
#include <hip/hip_bf16.h>

typedef __attribute__((ext_vector_type(16))) _Float16 v16h;
typedef __attribute__((ext_vector_type(8)))  _Float16 v8h;
typedef __attribute__((ext_vector_type(8)))  float    v8f;

#define NWG   64
#define ROWS  32          // batch rows per workgroup (2048 / 64)
#define TSTEP 64
#define BPTOT 2048
#define EPSV  1e-5f

// ---- workspace layout (bytes) ----
#define OFF_W0T 0
#define SZ_W0T  (256*160*2)                 // W0 transposed, f16, padded K=160
#define OFF_WST (OFF_W0T + SZ_W0T)
#define SZ_WST  (8*256*128*2)               // Ws transposed, f16
#define OFF_RED (OFF_WST + SZ_WST)
#define SZ_RED  (TSTEP*10*1024*4)           // 640 reduction slots x (512 sum + 512 sumsq)
#define OFF_BAR (OFF_RED + SZ_RED)          // 2 x u32 barrier state

// ---- LDS layout (floats) ----
// hacc:   8192   (h carry, then mean-accumulator -> next h; lifetimes disjoint)
// slots:  5*8192 (states s0,s1,s2,s3,s5 needed as GENO predecessors)
// c_raw:  16384  (64x256 raw GEMM output)
// axu:    8192   (A staging f16 64x160  UNION  pre-BN s 32x256 f32)
// redl:   1024   (per-layer mean / rstd)
// xbuf:   2*768  (double-buffered async x tile: 32 rows x 24 ch)
#define LDS_FLOATS (8192*6 + 16384 + 8192 + 1024 + 2*768)

__device__ __forceinline__ void grid_sync(unsigned* bar, unsigned* gen)
{
    __syncthreads();
    if (threadIdx.x == 0) {
        unsigned target = ++(*gen);
        __threadfence();
        unsigned old = atomicAdd(&bar[0], 1u);
        if (old == NWG - 1u) {
            atomicExch(&bar[0], 0u);
            __threadfence();
            atomicAdd(&bar[1], 1u);
        } else {
            while (atomicAdd(&bar[1], 0u) < target)
                __builtin_amdgcn_s_sleep(2);
        }
    }
    __syncthreads();
}

// address of x element feeding xt channel kk, row rglobal, scan position t
__device__ __forceinline__ const float* x_addr(const float* __restrict__ x,
                                               int rglobal, int t, int kk, bool rev)
{
    int hp = rglobal >> 5;
    int b  = rglobal & 31;
    int c  = kk >> 2;
    int dh = (kk >> 1) & 1;
    int dw = kk & 1;
    int w  = rev ? (63 - t) : t;
    return &x[((b * 3 + c) * 128 + (2 * hp + dh)) * 128 + (2 * w + dw)];
}

// ---- CDNA5 async global->LDS (per-lane gather DMA, tracked by ASYNCcnt) ----
__device__ __forceinline__ void async_ld_f32_to_lds(const float* gsrc, float* ldst)
{
    unsigned loff = (unsigned)(unsigned long long)ldst;   // generic LDS ptr: low 32 = LDS offset
    asm volatile("global_load_async_to_lds_b32 %0, %1, off"
                 :: "v"(loff), "v"(gsrc) : "memory");
}
__device__ __forceinline__ void wait_async0()
{
    asm volatile("s_wait_asynccnt 0x0" ::: "memory");
}

// issue the x gather for timestep t into xb (32 rows x [12 fwd | 12 rev])
__device__ __forceinline__ void issue_x_prefetch(const float* __restrict__ x,
                                                 float* __restrict__ xb,
                                                 int r0, int t, int tid)
{
    #pragma unroll
    for (int j = tid; j < ROWS * 24; j += 256) {
        const int  r   = j / 24;
        const int  c   = j % 24;
        const bool rev = c >= 12;
        const int  cc  = rev ? c - 12 : c;
        async_ld_f32_to_lds(x_addr(x, r0 + r, t, cc, rev), &xb[j]);
    }
}

__device__ __forceinline__ float apply_act(int id, float v)
{
    switch (id) {
        case 0:  return 1.f / (1.f + expf(-v));     // sigmoid
        case 1:  return v > 0.f ? v : 0.f;          // relu
        case 2:  return v;                          // identity
        default: return tanhf(v);                   // tanh
    }
}

// Fused bidirectional GEMM: C(64x256) = A(64xAK) * Bt(256xAK), f16 in, f32 out.
// Each wave owns one 16-row M tile (mt = wave>>1) and 8 N tiles; A fragments are
// loaded from LDS ONCE per wave and reused across all 8 N tiles (registers).
template <int AK, int KCH>
__device__ __forceinline__ void gemm64(const _Float16* __restrict__ a_lds,
                                       const _Float16* __restrict__ Bt,
                                       float* __restrict__ c_raw, int tid)
{
    const int wave   = tid >> 5;
    const int lane   = tid & 31;
    const int laneHi = lane >> 4;       // high half-wave holds K+8 / K+24
    const int lm     = lane & 15;
    const int mt     = wave >> 1;       // 4 M-tiles, 2 waves each
    const int ntBase = (wave & 1) * 8;  // 8 N-tiles per wave

    // hoist A fragments: identical for every N tile of this wave
    v16h afrag[KCH];
    {
        const _Float16* arow = a_lds + (mt * 16 + lm) * AK + laneHi * 8;
        #pragma unroll
        for (int kc = 0; kc < KCH; ++kc) {
            v8h a0 = *(const v8h*)(arow + kc * 32);
            v8h a1 = *(const v8h*)(arow + kc * 32 + 16);
            #pragma unroll
            for (int i = 0; i < 8; ++i) { afrag[kc][i] = a0[i]; afrag[kc][8 + i] = a1[i]; }
        }
    }

    #pragma unroll
    for (int tn = 0; tn < 8; ++tn) {
        const int nt = ntBase + tn;
        const _Float16* brow = Bt + (nt * 16 + lm) * AK + laneHi * 8;
        __builtin_prefetch(brow, 0, 1);
        v8f c = {};
        #pragma unroll
        for (int kc = 0; kc < KCH; ++kc) {
            v8h b0 = *(const v8h*)(brow + kc * 32);
            v8h b1 = *(const v8h*)(brow + kc * 32 + 16);
            v16h bv;
            #pragma unroll
            for (int i = 0; i < 8; ++i) { bv[i] = b0[i]; bv[8 + i] = b1[i]; }
            c = __builtin_amdgcn_wmma_f32_16x16x32_f16(
                    false, afrag[kc], false, bv, (short)0, c, false, false);
        }
        #pragma unroll
        for (int i = 0; i < 8; ++i)
            c_raw[(mt * 16 + laneHi * 8 + i) * 256 + nt * 16 + lm] = c[i];
    }
}

// Batch-norm over the full 2048-row batch for a 32x256 per-WG tile.
// accmode: 0 = none, 1 = acc = v, 2 = acc += v.
__device__ __forceinline__ void bn_state(const float* __restrict__ s_pre,
                                         float* __restrict__ redslot,
                                         float* __restrict__ redl,
                                         const float* __restrict__ gamma,
                                         const float* __restrict__ beta,
                                         float* __restrict__ dst,
                                         float* __restrict__ hacc,
                                         int accmode,
                                         unsigned* bar, unsigned* gen, int tid)
{
    {   // per-WG partial sums (256 threads <-> 256 channels)
        const int ch = tid;
        float s = 0.f, s2 = 0.f;
        #pragma unroll 4
        for (int r = 0; r < ROWS; ++r) {
            float v = s_pre[r * 256 + ch];
            s += v; s2 += v * v;
        }
        atomicAdd(&redslot[ch], s);
        atomicAdd(&redslot[512 + ch], s2);
    }
    grid_sync(bar, gen);
    {
        const int ch = tid;
        float m   = redslot[ch] * (1.f / (float)BPTOT);
        float var = redslot[512 + ch] * (1.f / (float)BPTOT) - m * m;
        redl[ch]       = m;
        redl[512 + ch] = rsqrtf(var + EPSV);
    }
    __syncthreads();
    for (int i = tid; i < ROWS * 256; i += 256) {
        const int ch = i & 255;
        float v = (s_pre[i] - redl[ch]) * redl[512 + ch] * gamma[ch] + beta[ch];
        if (dst) dst[i] = v;
        if (accmode == 1)      hacc[i]  = v;
        else if (accmode == 2) hacc[i] += v;
    }
    __syncthreads();
}

__global__ void zero_scratch_kernel(float* red, unsigned n, unsigned* bar)
{
    unsigned i = blockIdx.x * blockDim.x + threadIdx.x;
    if (i < n) red[i] = 0.f;
    if (i < 2) bar[i] = 0u;
}

__global__ void prep_weights_kernel(const float* __restrict__ W0,
                                    const float* __restrict__ Ws,
                                    _Float16* __restrict__ W0t,
                                    _Float16* __restrict__ Wst)
{
    int i = blockIdx.x * blockDim.x + threadIdx.x;
    if (i < 256 * 160) {                 // W0t[n][k] = W0[k][n], zero-pad k>=140
        int n = i / 160, k = i % 160;
        W0t[i] = (_Float16)(k < 140 ? W0[k * 256 + n] : 0.f);
    }
    if (i < 8 * 256 * 128) {             // Wst[w][n][k] = Ws[w][k][n]
        int w = i / (256 * 128);
        int rem = i % (256 * 128);
        int n = rem / 128, k = rem % 128;
        Wst[i] = (_Float16)Ws[(w * 128 + k) * 256 + n];
    }
}

__global__ void __launch_bounds__(256)
rnn_scan_kernel(const float* __restrict__ x,
                const float* __restrict__ bn_in_g,  const float* __restrict__ bn_in_b,
                const float* __restrict__ bn_rev_g, const float* __restrict__ bn_rev_b,
                const float* __restrict__ bn_g,     const float* __restrict__ bn_b,
                const _Float16* __restrict__ W0t,
                const _Float16* __restrict__ Wst,
                float* __restrict__ red,
                unsigned* __restrict__ bar,
                float* __restrict__ out)
{
    extern __shared__ float smem[];
    float*    hacc  = smem;                         // 8192 (h then accumulator)
    float*    slots = smem + 8192;                  // 5 * 8192
    float*    c_raw = smem + 8192 * 6;              // 16384
    float*    axu   = smem + 8192 * 6 + 16384;      // 8192 (union)
    float*    redl  = axu + 8192;                   // 1024
    float*    xbufs = redl + 1024;                  // 2 * 768
    _Float16* a_lds = (_Float16*)axu;
    float*    s_pre = axu;

    const int tid = threadIdx.x;
    const int r0  = blockIdx.x * ROWS;
    unsigned  gen = 0;

    // GENO wiring (state index -> slot: 0->0, 1->1, 2->2, 3->3, 5->4)
    const int PRED_SLOT[8] = {0, 1, 1, 1, 2, 4, 3, 4};
    const int DST_SLOT[8]  = {1, 2, 3, -1, 4, -1, -1, -1};
    const int ACT_ID[8]    = {0, 1, 1, 2, 3, 0, 3, 1};  // sig,relu,relu,id,tanh,sig,tanh,relu

    for (int i = tid; i < 8192; i += 256) hacc[i] = 0.f;   // h0 = 0
    __syncthreads();

    issue_x_prefetch(x, xbufs, r0, 0, tid);                // t=0 x tile in flight

    for (int t = 0; t < TSTEP; ++t) {
        float* xb  = xbufs + (t & 1) * 768;
        float* xbn = xbufs + ((t + 1) & 1) * 768;

        wait_async0();          // this wave's async gather done (LDS written)
        __syncthreads();        // visible to all waves

        if (t + 1 < TSTEP)      // overlap next x gather with whole step body
            issue_x_prefetch(x, xbn, r0, t + 1, tid);

        // ---------- input layer: BN(concat(x_t, h)) over batch, fwd + rev ----------
        float* rslot = red + (t * 10 + 0) * 1024;
        for (int ch = tid; ch < 280; ch += 256) {
            const bool rev = ch >= 140;
            const int  cc  = rev ? ch - 140 : ch;
            float s = 0.f, s2 = 0.f;
            for (int r = 0; r < ROWS; ++r) {
                float v = (cc < 12) ? xb[r * 24 + (rev ? 12 : 0) + cc]
                                    : hacc[r * 256 + (rev ? 128 : 0) + cc - 12];
                s += v; s2 += v * v;
            }
            atomicAdd(&rslot[ch], s);
            atomicAdd(&rslot[512 + ch], s2);
        }
        grid_sync(bar, &gen);
        for (int ch = tid; ch < 280; ch += 256) {
            float m   = rslot[ch] * (1.f / (float)BPTOT);
            float var = rslot[512 + ch] * (1.f / (float)BPTOT) - m * m;
            redl[ch]       = m;
            redl[512 + ch] = rsqrtf(var + EPSV);
        }
        __syncthreads();

        // build normalized A (64 x 160 f16): rows 0..31 fwd, 32..63 rev
        for (int i = tid; i < 64 * 160; i += 256) {
            const int mm = i / 160, kk = i % 160;
            _Float16 hv = (_Float16)0.f;
            if (kk < 140) {
                const bool rev = mm >= 32;
                const int  r   = rev ? mm - 32 : mm;
                float v = (kk < 12) ? xb[r * 24 + (rev ? 12 : 0) + kk]
                                    : hacc[r * 256 + (rev ? 128 : 0) + kk - 12];
                const int rc = (rev ? 140 : 0) + kk;
                float g = rev ? bn_rev_g[kk] : bn_in_g[kk];
                float b = rev ? bn_rev_b[kk] : bn_in_b[kk];
                hv = (_Float16)((v - redl[rc]) * redl[512 + rc] * g + b);
            }
            a_lds[i] = hv;
        }
        __syncthreads();

        gemm64<160, 5>(a_lds, W0t, c_raw, tid);
        __syncthreads();

        // gated update: s = h + sigmoid(gate) * (tanh(cand) - h)   (a_lds dead -> s_pre)
        for (int i = tid; i < 8192; i += 256) {
            const int r = i >> 8, ch = i & 255;
            float gate, cand;
            if (ch < 128) { gate = c_raw[r * 256 + ch];             cand = c_raw[r * 256 + 128 + ch]; }
            else          { gate = c_raw[(32 + r) * 256 + ch - 128]; cand = c_raw[(32 + r) * 256 + ch]; }
            float hp = hacc[i];
            float sg = 1.f / (1.f + expf(-gate));
            s_pre[i] = hp + sg * (tanhf(cand) - hp);
        }
        __syncthreads();

        // s0 = BN(...) -> slot0 (not in the mean)
        bn_state(s_pre, red + (t * 10 + 1) * 1024, redl,
                 bn_g + 0 * 256, bn_b + 0 * 256,
                 slots + 0 * 8192, hacc, /*accmode=*/0, bar, &gen, tid);

        // ---------- GENO layers ----------
        for (int gi = 0; gi < 8; ++gi) {
            float* sp = slots + PRED_SLOT[gi] * 8192;

            // A (64 x 128 f16) from predecessor state (fwd rows then rev rows)
            for (int i = tid; i < 64 * 128; i += 256) {
                const int mm = i >> 7, kk = i & 127;
                const bool rev = mm >= 32;
                const int  r   = rev ? mm - 32 : mm;
                a_lds[i] = (_Float16)sp[r * 256 + (rev ? 128 : 0) + kk];
            }
            __syncthreads();

            gemm64<128, 4>(a_lds, Wst + gi * 256 * 128, c_raw, tid);
            __syncthreads();

            const int act = ACT_ID[gi];
            for (int i = tid; i < 8192; i += 256) {
                const int r = i >> 8, ch = i & 255;
                float gate, cand;
                if (ch < 128) { gate = c_raw[r * 256 + ch];             cand = c_raw[r * 256 + 128 + ch]; }
                else          { gate = c_raw[(32 + r) * 256 + ch - 128]; cand = c_raw[(32 + r) * 256 + ch]; }
                float hp = sp[i];
                float sg = 1.f / (1.f + expf(-gate));
                s_pre[i] = hp + sg * (apply_act(act, cand) - hp);
            }
            __syncthreads();

            const int ds = DST_SLOT[gi];
            bn_state(s_pre, red + (t * 10 + 2 + gi) * 1024, redl,
                     bn_g + (gi + 1) * 256, bn_b + (gi + 1) * 256,
                     ds >= 0 ? slots + ds * 8192 : (float*)nullptr,
                     hacc, gi == 0 ? 1 : 2, bar, &gen, tid);
        }

        // out = mean(states 1..8); becomes h for next step. out[b, ch, hp, t]
        for (int i = tid; i < 8192; i += 256) {
            float v = hacc[i] * 0.125f;
            hacc[i] = v;
            const int r = i >> 8, ch = i & 255;
            const int rg = r0 + r;
            const int hp = rg >> 5, b = rg & 31;
            out[((b * 256 + ch) * 64 + hp) * 64 + t] = v;
        }
        __syncthreads();
    }
}

extern "C" void kernel_launch(void* const* d_in, const int* in_sizes, int n_in,
                              void* d_out, int out_size, void* d_ws, size_t ws_size,
                              hipStream_t stream)
{
    const float* x        = (const float*)d_in[0];
    const float* W0       = (const float*)d_in[1];
    const float* Ws       = (const float*)d_in[2];
    const float* bn_in_g  = (const float*)d_in[3];
    const float* bn_in_b  = (const float*)d_in[4];
    const float* bn_rev_g = (const float*)d_in[5];
    const float* bn_rev_b = (const float*)d_in[6];
    const float* bn_g     = (const float*)d_in[7];
    const float* bn_b     = (const float*)d_in[8];
    float* out = (float*)d_out;

    char* ws = (char*)d_ws;
    _Float16* W0t = (_Float16*)(ws + OFF_W0T);
    _Float16* Wst = (_Float16*)(ws + OFF_WST);
    float*    red = (float*)(ws + OFF_RED);
    unsigned* bar = (unsigned*)(ws + OFF_BAR);

    const unsigned nred = TSTEP * 10 * 1024;
    zero_scratch_kernel<<<(nred + 255) / 256, 256, 0, stream>>>(red, nred, bar);
    prep_weights_kernel<<<(8 * 256 * 128 + 255) / 256, 256, 0, stream>>>(W0, Ws, W0t, Wst);

    const size_t lds_bytes = (size_t)LDS_FLOATS * sizeof(float);  // 305152 B < 320 KB
    rnn_scan_kernel<<<NWG, 256, lds_bytes, stream>>>(
        x, bn_in_g, bn_in_b, bn_rev_g, bn_rev_b, bn_g, bn_b,
        W0t, Wst, red, bar, out);
}
